// MixtureOfExperts_28355374088232
// MI455X (gfx1250) — matrix operand order repaired and checked by reading, not compiled, not run on hardware
//
#include <hip/hip_runtime.h>

typedef __attribute__((ext_vector_type(16))) _Float16 v16h;
typedef __attribute__((ext_vector_type(8)))  float    v8f;

#define NTILE 43            // 2 gate1 + 1 gate2 + 8*(2 W1 + 2 W2 + 1 W3)
#define REC_U32 16          // per-lane record: 8 u32 (16 f16 A) + 8 f32 bias
#define LDS_U32 (NTILE * 32 * REC_U32)

__device__ inline unsigned packpair(float lo, float hi) {
    union { _Float16 h[2]; unsigned u; } cv;
    cv.h[0] = (_Float16)lo; cv.h[1] = (_Float16)hi;
    return cv.u;
}

// cvt_pk canonicalizes, so a bare packed max against 0 is a correct relu:
// 2 VALU ops per dword pair, no v_max canonicalize pairs.
__device__ inline unsigned cvt_pk_relu(float lo, float hi) {
    unsigned r;
    asm("v_cvt_pk_f16_f32 %0, %1, %2\n\t"
        "v_pk_max_num_f16 %0, %0, 0"
        : "=v"(r) : "v"(lo), "v"(hi));
    return r;
}

__device__ inline float fmax_(float a, float b) {
    float r;
    asm("v_max_num_f32 %0, %1, %2" : "=v"(r) : "v"(a), "v"(b));
    return r;
}

__device__ inline v16h ldsA(const unsigned* smem, int t, int lane) {
    union { uint4 u[2]; v16h v; } cv;
    const uint4* p = (const uint4*)(smem + ((t * 32 + lane) << 4));
    cv.u[0] = p[0]; cv.u[1] = p[1];
    return cv.v;
}

__device__ inline v8f ldsBias(const unsigned* smem, int t, int lane) {
    union { float4 f[2]; v8f v; } cv;
    const float4* p = (const float4*)(smem + ((t * 32 + lane) << 4) + 8);
    cv.f[0] = p[0]; cv.f[1] = p[1];
    return cv.v;
}

__device__ inline v8f wmma_(v16h a, v16h b, v8f c) {
    // D(16x16,f32) = A(16x32,f16) x B(32x16,f16) + C ; bias pre-loaded into C
    return __builtin_amdgcn_wmma_f32_16x16x32_f16(
        false, a, false, b, (short)0, c, false, false);
}

__device__ inline v16h packrelu(v8f clo, v8f chi) {
    // C0 -> B slots 0..7, C1 -> B slots 8..15 (lane-local); dword j = halves (2j, 2j+1)
    union { unsigned u[8]; v16h v; } r;
#pragma unroll
    for (int i = 0; i < 4; ++i) {
        r.u[i]     = cvt_pk_relu(clo[2 * i], clo[2 * i + 1]);
        r.u[4 + i] = cvt_pk_relu(chi[2 * i], chi[2 * i + 1]);
    }
    return r.v;
}

// 32-output layer: two A tiles (t, t+1), two token tiles in flight
__device__ inline void layer32(const unsigned* smem, int t, int lane,
                               v16h b0, v16h b1, v16h& o0, v16h& o1) {
    v16h a0 = ldsA(smem, t, lane);
    v16h a1 = ldsA(smem, t + 1, lane);
    v8f bi0 = ldsBias(smem, t, lane);
    v8f bi1 = ldsBias(smem, t + 1, lane);
    v8f c00 = wmma_(a0, b0, bi0);
    v8f c10 = wmma_(a1, b0, bi1);
    v8f c01 = wmma_(a0, b1, bi0);
    v8f c11 = wmma_(a1, b1, bi1);
    o0 = packrelu(c00, c10);
    o1 = packrelu(c01, c11);
}

__device__ inline void softmax8(v8f l, float g[8]) {
    float mx = fmax_(l[0], l[1]);
    mx = fmax_(mx, fmax_(l[2], l[3]));
    mx = fmax_(mx, fmax_(l[4], l[5]));
    mx = fmax_(mx, fmax_(l[6], l[7]));
    float s = 0.f;
#pragma unroll
    for (int i = 0; i < 8; ++i) { float e = __expf(l[i] - mx); g[i] = e; s += e; }
    float inv = __frcp_rn(s);
#pragma unroll
    for (int i = 0; i < 8; ++i) g[i] *= inv;
}

__device__ inline v16h loadX(const float* __restrict__ x, long base, int mrow,
                             int hi, int Btotal) {
    v16h r;
#pragma unroll
    for (int i = 0; i < 16; ++i) r[i] = (_Float16)0.f;
    long tok = base + mrow;
    if (hi == 0 && tok < (long)Btotal) {  // K slots 0..5 = features, rest zero-pad
        const float2* p = (const float2*)(x + tok * 6);
        float2 q0 = p[0], q1 = p[1], q2 = p[2];
        r[0] = (_Float16)q0.x; r[1] = (_Float16)q0.y;
        r[2] = (_Float16)q1.x; r[3] = (_Float16)q1.y;
        r[4] = (_Float16)q2.x; r[5] = (_Float16)q2.y;
    }
    return r;
}

__global__ __launch_bounds__(256) void MixtureOfExperts_28355374088232_kernel(
    const float* __restrict__ x,
    const float* __restrict__ W1,  const float* __restrict__ b1,
    const float* __restrict__ W2,  const float* __restrict__ b2,
    const float* __restrict__ W3,  const float* __restrict__ b3,
    const float* __restrict__ Wg1, const float* __restrict__ bg1,
    const float* __restrict__ Wg2, const float* __restrict__ bg2,
    float* __restrict__ out, int Btotal) {
    extern __shared__ unsigned smem[];

    // ---- Stage 1: build f16 A-tiles (Wt, feature-permuted rows) + f32 bias records in LDS
    for (int rec = threadIdx.x; rec < NTILE * 32; rec += blockDim.x) {
        int t = rec >> 5, l = rec & 31;
        int mrow = l & 15, hi = l >> 4;
        const float* Wp; const float* bp; int Kreal, Mcols, Mreal, Tt;
        if (t < 2)       { Wp = Wg1; bp = bg1; Kreal = 6;  Mcols = 32; Mreal = 32; Tt = t; }
        else if (t == 2) { Wp = Wg2; bp = bg2; Kreal = 32; Mcols = 8;  Mreal = 8;  Tt = 0; }
        else {
            int u = t - 3, e = u / 5, s = u % 5;
            if (s < 2)      { Wp = W1 + e * 192;  bp = b1 + e * 32; Kreal = 6;  Mcols = 32; Mreal = 32; Tt = s;     }
            else if (s < 4) { Wp = W2 + e * 1024; bp = b2 + e * 32; Kreal = 32; Mcols = 32; Mreal = 32; Tt = s - 2; }
            else            { Wp = W3 + e * 64;   bp = b3 + e * 2;  Kreal = 32; Mcols = 2;  Mreal = 2;  Tt = 0;     }
        }
        unsigned* dst = smem + rec * REC_U32;
        // row -> output-feature permutation: {0..7,16..23} / {8..15,24..31} for 32-wide layers
        int fout = (Mreal == 32) ? ((mrow < 8 ? mrow : mrow + 8) + 8 * Tt) : mrow;
        bool mv = fout < Mreal;
#pragma unroll
        for (int j = 0; j < 8; ++j) {  // A slots: lane<16 -> K{0..15}; lane>=16 -> K{16..31}
            int s0 = 2 * j, s1 = 2 * j + 1;
            int k0 = s0 + 16 * hi;
            int k1 = s1 + 16 * hi;
            float v0 = (mv && k0 < Kreal) ? Wp[k0 * Mcols + fout] : 0.f;
            float v1 = (mv && k1 < Kreal) ? Wp[k1 * Mcols + fout] : 0.f;
            dst[j] = packpair(v0, v1);
        }
#pragma unroll
        for (int r = 0; r < 8; ++r) {  // bias in C layout: row = r + 8*hi
            int row = r + 8 * hi;
            int fb = (Mreal == 32) ? ((row < 8 ? row : row + 8) + 8 * Tt) : row;
            ((float*)dst)[8 + r] = (fb < Mreal) ? bp[fb] : 0.f;
        }
    }
    __syncthreads();

    // ---- Stage 2: grid-stride over pairs of 16-token tiles
    int lane = threadIdx.x & 31;
    int mrow = lane & 15, hi = lane >> 4;
    int wave = blockIdx.x * (blockDim.x >> 5) + (threadIdx.x >> 5);
    int nw   = gridDim.x * (blockDim.x >> 5);
    int ntiles = (Btotal + 15) >> 4;
    int npairs = (ntiles + 1) >> 1;

#pragma unroll 1
    for (int p = wave; p < npairs; p += nw) {
        // Opaque memory barrier: stops LICM from hoisting the (loop-invariant)
        // LDS weight-tile loads out of the loop, which would need ~690 VGPRs
        // and spill to scratch. Forces per-iteration ds_load streaming instead.
        asm volatile("" ::: "memory");

        long base0 = (long)(2 * p) * 16;
        long base1 = base0 + 16;
        v16h xb0 = loadX(x, base0, mrow, hi, Btotal);
        v16h xb1 = loadX(x, base1, mrow, hi, Btotal);

        // gate: relu(x@Wg1+bg1) @ Wg2 + bg2 -> softmax (lane-local 8 logits)
        v16h gb0, gb1;
        layer32(smem, 0, lane, xb0, xb1, gb0, gb1);
        v16h a2 = ldsA(smem, 2, lane);
        v8f bi2 = ldsBias(smem, 2, lane);
        v8f l0 = wmma_(a2, gb0, bi2);
        v8f l1 = wmma_(a2, gb1, bi2);
        float g0[8], g1[8];
        softmax8(l0, g0);
        softmax8(l1, g1);

        float a00 = 0.f, a01 = 0.f, a10 = 0.f, a11 = 0.f;
#pragma unroll
        for (int e = 0; e < 8; ++e) {
            int tb = 3 + e * 5;
            v16h h0, h1;
            layer32(smem, tb,     lane, xb0, xb1, h0, h1);   // h1 = relu(x@W1+b1)
            layer32(smem, tb + 2, lane, h0,  h1,  h0, h1);   // h2 = relu(h1@W2+b2)
            v16h a3 = ldsA(smem, tb + 4, lane);              // y = h2@W3+b3 (rows 0,1)
            v8f bi3 = ldsBias(smem, tb + 4, lane);
            v8f y0 = wmma_(a3, h0, bi3);
            v8f y1 = wmma_(a3, h1, bi3);
            a00 += g0[e] * y0[0]; a01 += g0[e] * y0[1];
            a10 += g1[e] * y1[0]; a11 += g1[e] * y1[1];
        }

        if (hi == 0) {
            long tok0 = base0 + mrow;
            if (tok0 < (long)Btotal) ((float2*)out)[tok0] = make_float2(a00, a01);
            long tok1 = base1 + mrow;
            if (tok1 < (long)Btotal) ((float2*)out)[tok1] = make_float2(a10, a11);
        }
    }
}

extern "C" void kernel_launch(void* const* d_in, const int* in_sizes, int n_in,
                              void* d_out, int out_size, void* d_ws, size_t ws_size,
                              hipStream_t stream) {
    const float* x   = (const float*)d_in[0];
    const float* W1  = (const float*)d_in[1];
    const float* b1  = (const float*)d_in[2];
    const float* W2  = (const float*)d_in[3];
    const float* b2  = (const float*)d_in[4];
    const float* W3  = (const float*)d_in[5];
    const float* b3  = (const float*)d_in[6];
    const float* Wg1 = (const float*)d_in[7];
    const float* bg1 = (const float*)d_in[8];
    const float* Wg2 = (const float*)d_in[9];
    const float* bg2 = (const float*)d_in[10];
    int Btotal = in_sizes[0] / 6;

    size_t shmem = (size_t)LDS_U32 * 4;  // 86 KB dynamic LDS (<= 320 KB/WGP)
    dim3 grid(1024), block(256);         // 8 wave32 per block; grid-stride over tile pairs
    hipLaunchKernelGGL(MixtureOfExperts_28355374088232_kernel, grid, block, shmem, stream,
                       x, W1, b1, W2, b2, W3, b3, Wg1, bg1, Wg2, bg2,
                       (float*)d_out, Btotal);
}
